// GRU_72387378806936
// MI455X (gfx1250) — matrix operand order repaired and checked by reading, not compiled, not run on hardware
//
#include <hip/hip_runtime.h>
#include <hip/hip_bf16.h>

typedef _Float16 v16h __attribute__((ext_vector_type(16)));
typedef _Float16 h8   __attribute__((ext_vector_type(8)));
typedef float    v8f  __attribute__((ext_vector_type(8)));

#define B_TOT  256
#define S_LEN  512
#define D_IN   64
#define H_DIM  256
#define MLP_HD 64
#define D_OUT  16

// Fragment-major weight storage: one 16x16x32 B-fragment = 32 lanes * 16 halves = 512 halves.
// Within a fragment, lane L holds column n = L%16, K-values k0 + (L/16)*8 + (e%8) + (e/8)*16 (e = 0..15),
// stored contiguously so load_b is a single 32-byte LDS read per lane.
struct Smem {
  _Float16 whh[128 * 512];   // W_hh^T frags: 16 ntiles x 8 ktiles   (128 KB)
  _Float16 wih[ 32 * 512];   // W_ih^T frags: 16 ntiles x 2 ktiles   ( 32 KB)
  _Float16 w1 [ 32 * 512];   // W1^T  frags:  4 ntiles x 8 ktiles    ( 32 KB)
  _Float16 w2 [  8 * 512];   // W2^T  frags:  4 ntiles x 2 ktiles    (  8 KB)
  _Float16 w3 [  2 * 512];   // W3^T  frags:  1 ntile  x 2 ktiles    (  2 KB)
  _Float16 h[2][16 * H_DIM]; // ping-pong hidden state               ( 16 KB)
  _Float16 xt[16 * D_IN];    // current (possibly spliced) input tile
  _Float16 z1[16 * MLP_HD];
  _Float16 z2[16 * MLP_HD];
  _Float16 last[16 * D_OUT]; // previous step's output (for splice)
  float bias_h[H_DIM];       // b_ih + b_hh
  float b1v[MLP_HD];
  float b2v[MLP_HD];
  float b3v[D_OUT];
};

// A-operand (16xK row-major f16 in LDS) -> v16h per the CDNA5 16-bit A layout:
// lanes 0-15: M=lane, halves 0..7 = K k0..k0+7, halves 8..15 = K k0+16..k0+23
// lanes 16-31: M=lane-16, halves 0..7 = K k0+8..k0+15, halves 8..15 = K k0+24..k0+31
__device__ inline v16h load_a(const _Float16* __restrict__ src, int ld, int k0, int lane) {
  const int m  = lane & 15;
  const int hi = lane >> 4;
  const _Float16* p = src + m * ld + k0 + hi * 8;
  h8 lo = *(const h8*)p;         // 16B LDS load
  h8 hb = *(const h8*)(p + 16);  // 16B LDS load
  return __builtin_shufflevector(lo, hb, 0,1,2,3,4,5,6,7,8,9,10,11,12,13,14,15);
}

__device__ inline v16h load_b(const _Float16* __restrict__ frag, int lane) {
  return *(const v16h*)(frag + lane * 16);  // 32B contiguous per lane
}

__device__ inline v8f wmma_f16(v16h a, v16h b, v8f c) {
  return __builtin_amdgcn_wmma_f32_16x16x32_f16(false, a, false, b, (short)0, c, false, false);
}

// Convert a row-major fp32 weight W[N][K] (we need B[k][n] = W[n][k]) into fragment-major f16.
__device__ inline void cvt_frags(_Float16* __restrict__ dst, const float* __restrict__ W,
                                 int ldK, int Ktiles, int nfrags, int tid, int nthr) {
  const int total = nfrags * 512;
  for (int idx = tid; idx < total; idx += nthr) {
    const int frag = idx >> 9;
    const int w    = idx & 511;
    const int lane = w >> 4;
    const int e    = w & 15;
    const int kt   = frag % Ktiles;
    const int nt   = frag / Ktiles;
    const int n    = nt * 16 + (lane & 15);
    const int k    = kt * 32 + (lane >> 4) * 8 + (e & 7) + (e >> 3) * 16;
    dst[idx] = (_Float16)W[n * ldK + k];
  }
}

__global__ __launch_bounds__(128, 1) void rnn_persistent(
    const float* __restrict__ x,
    const float* __restrict__ W_ih, const float* __restrict__ b_ih,
    const float* __restrict__ W_hh, const float* __restrict__ b_hh,
    const float* __restrict__ W1, const float* __restrict__ b1,
    const float* __restrict__ W2, const float* __restrict__ b2,
    const float* __restrict__ W3, const float* __restrict__ b3,
    const int* __restrict__ pred_start_p,
    float* __restrict__ out)
{
  extern __shared__ __align__(32) char smem_raw[];
  Smem& sm = *reinterpret_cast<Smem*>(smem_raw);

  const int tid  = threadIdx.x;
  const int nthr = blockDim.x;
  const int lane = tid & 31;
  const int wid  = tid >> 5;        // wave id, 0..3 (wave32)
  const int b0   = blockIdx.x * 16; // batch tile base
  const int ps   = *pred_start_p;

  // ---- one-time: stage all weights into LDS (fragment-major f16) ----
  cvt_frags(sm.whh, W_hh, H_DIM,  8, 128, tid, nthr);
  cvt_frags(sm.wih, W_ih, D_IN,   2,  32, tid, nthr);
  cvt_frags(sm.w1,  W1,   H_DIM,  8,  32, tid, nthr);
  cvt_frags(sm.w2,  W2,   MLP_HD, 2,   8, tid, nthr);
  cvt_frags(sm.w3,  W3,   MLP_HD, 2,   2, tid, nthr);
  for (int i = tid; i < H_DIM;  i += nthr) sm.bias_h[i] = b_ih[i] + b_hh[i];
  for (int i = tid; i < MLP_HD; i += nthr) { sm.b1v[i] = b1[i]; sm.b2v[i] = b2[i]; }
  for (int i = tid; i < D_OUT;  i += nthr) sm.b3v[i] = b3[i];
  for (int i = tid; i < 16 * H_DIM; i += nthr) sm.h[0][i] = (_Float16)0.f;
  for (int i = tid; i < 16 * D_OUT; i += nthr) sm.last[i] = (_Float16)0.f;
  __syncthreads();

  const int n_lo = lane & 15;
  const int hi   = lane >> 4;
  const int m_x  = tid >> 3;        // xt-build row      (128 threads -> 16 rows x 8 blocks)
  const int kblk = (tid & 7) * 8;   // xt-build col block

  int cur = 0;
  for (int t = 0; t < S_LEN; ++t) {
    // ---- 1) build xt tile (with autoregressive splice into first D_OUT cols) ----
    if (kblk < D_OUT && t > ps) {
      _Float16* d = sm.xt + m_x * D_IN + kblk;
      #pragma unroll
      for (int i = 0; i < 8; ++i) d[i] = sm.last[m_x * D_OUT + kblk + i];
    } else {
      const float* xp = x + (((size_t)(b0 + m_x) * S_LEN) + t) * D_IN + kblk;
      float4 u0 = *(const float4*)xp;
      float4 u1 = *(const float4*)(xp + 4);
      _Float16* d = sm.xt + m_x * D_IN + kblk;
      d[0]=(_Float16)u0.x; d[1]=(_Float16)u0.y; d[2]=(_Float16)u0.z; d[3]=(_Float16)u0.w;
      d[4]=(_Float16)u1.x; d[5]=(_Float16)u1.y; d[6]=(_Float16)u1.z; d[7]=(_Float16)u1.w;
    }
    __syncthreads();

    // ---- 2) recurrent GEMM: h' = tanh(xt @ Wih^T + h @ Whh^T + bias), 4 N-tiles/wave ----
    {
      const _Float16* hprev = sm.h[cur];
      _Float16*       hnext = sm.h[cur ^ 1];
      const int ntb = wid * 4;
      v8f acc[4] = {};
      #pragma unroll
      for (int kt = 0; kt < 2; ++kt) {
        v16h a = load_a(sm.xt, D_IN, kt * 32, lane);
        #pragma unroll
        for (int j = 0; j < 4; ++j)
          acc[j] = wmma_f16(a, load_b(sm.wih + (((ntb + j) * 2) + kt) * 512, lane), acc[j]);
      }
      #pragma unroll
      for (int kt = 0; kt < 8; ++kt) {
        v16h a = load_a(hprev, H_DIM, kt * 32, lane);
        #pragma unroll
        for (int j = 0; j < 4; ++j)
          acc[j] = wmma_f16(a, load_b(sm.whh + (((ntb + j) * 8) + kt) * 512, lane), acc[j]);
      }
      #pragma unroll
      for (int j = 0; j < 4; ++j) {
        const int n = (ntb + j) * 16 + n_lo;
        const float bias = sm.bias_h[n];
        #pragma unroll
        for (int v = 0; v < 8; ++v) {
          float y = tanhf(acc[j][v] + bias);
          hnext[(v + 8 * hi) * H_DIM + n] = (_Float16)y;   // C-layout: M = v + 8*(lane/16)
        }
      }
    }
    __syncthreads();

    // ---- 3) MLP layer 1: z1 = relu(h' @ W1^T + b1), 1 N-tile/wave ----
    {
      const _Float16* hn = sm.h[cur ^ 1];
      v8f acc = {};
      #pragma unroll
      for (int kt = 0; kt < 8; ++kt)
        acc = wmma_f16(load_a(hn, H_DIM, kt * 32, lane),
                       load_b(sm.w1 + ((wid * 8) + kt) * 512, lane), acc);
      const int n = wid * 16 + n_lo;
      const float bias = sm.b1v[n];
      #pragma unroll
      for (int v = 0; v < 8; ++v) {
        float y = fmaxf(acc[v] + bias, 0.f);
        sm.z1[(v + 8 * hi) * MLP_HD + n] = (_Float16)y;
      }
    }
    __syncthreads();

    // ---- 4) MLP layer 2: z2 = relu(z1 @ W2^T + b2), 1 N-tile/wave ----
    {
      v8f acc = {};
      #pragma unroll
      for (int kt = 0; kt < 2; ++kt)
        acc = wmma_f16(load_a(sm.z1, MLP_HD, kt * 32, lane),
                       load_b(sm.w2 + ((wid * 2) + kt) * 512, lane), acc);
      const int n = wid * 16 + n_lo;
      const float bias = sm.b2v[n];
      #pragma unroll
      for (int v = 0; v < 8; ++v) {
        float y = fmaxf(acc[v] + bias, 0.f);
        sm.z2[(v + 8 * hi) * MLP_HD + n] = (_Float16)y;
      }
    }
    __syncthreads();

    // ---- 5) MLP layer 3 (wave 0 only): o = z2 @ W3^T + b3 -> global out + splice buffer ----
    if (wid == 0) {
      v8f acc = {};
      #pragma unroll
      for (int kt = 0; kt < 2; ++kt)
        acc = wmma_f16(load_a(sm.z2, MLP_HD, kt * 32, lane),
                       load_b(sm.w3 + kt * 512, lane), acc);
      const float bias = sm.b3v[n_lo];
      #pragma unroll
      for (int v = 0; v < 8; ++v) {
        const int m = v + 8 * hi;
        const float y = acc[v] + bias;
        out[(((size_t)(b0 + m) * S_LEN) + t) * D_OUT + n_lo] = y;
        sm.last[m * D_OUT + n_lo] = (_Float16)y;
      }
    }
    __syncthreads();

    cur ^= 1;
  }
}

extern "C" void kernel_launch(void* const* d_in, const int* in_sizes, int n_in,
                              void* d_out, int out_size, void* d_ws, size_t ws_size,
                              hipStream_t stream) {
  const float* x    = (const float*)d_in[0];
  const float* W_ih = (const float*)d_in[1];
  const float* b_ih = (const float*)d_in[2];
  const float* W_hh = (const float*)d_in[3];
  const float* b_hh = (const float*)d_in[4];
  const float* W1   = (const float*)d_in[5];
  const float* b1   = (const float*)d_in[6];
  const float* W2   = (const float*)d_in[7];
  const float* b2   = (const float*)d_in[8];
  const float* W3   = (const float*)d_in[9];
  const float* b3   = (const float*)d_in[10];
  const int*   ps   = (const int*)d_in[11];
  float* out = (float*)d_out;

  static_assert(sizeof(Smem) <= 320 * 1024, "LDS budget exceeded");
  const size_t shmem = sizeof(Smem);
  hipFuncSetAttribute(reinterpret_cast<const void*>(rnn_persistent),
                      hipFuncAttributeMaxDynamicSharedMemorySize, (int)shmem);

  rnn_persistent<<<B_TOT / 16, 128, shmem, stream>>>(
      x, W_ih, b_ih, W_hh, b_hh, W1, b1, W2, b2, W3, b3, ps, out);
}
